// TemporalTransformer_83597243449591
// MI455X (gfx1250) — compile-verified, matrix operand order checked by reading, not run on hardware
//
#include <hip/hip_runtime.h>
#include <hip/hip_bf16.h>
#include <math.h>

// ---------------------------------------------------------------------------
// Problem constants (from the reference)
// ---------------------------------------------------------------------------
#define TT      16
#define CC      512
#define SS      144           // H*W
#define NN      2304          // T*S
#define NHEADS  16
#define DH      64
#define INNER   1024
#define MLP     2048
#define MAXT    32
#define NTQ     144           // NN/16 query tiles
#define ATT_SCALE 0.125f      // DH^-0.5

typedef _Float16 h16;
typedef __attribute__((ext_vector_type(16))) _Float16 v16h;
typedef __attribute__((ext_vector_type(8)))  float    v8f;

// K index held by (lane, element i) of a 16-bit A/B WMMA fragment
// (ISA 7.12.2: VGPR v holds K pair {2(v%4) + 16(v/4) + 8*(lane/16)} )
__device__ __forceinline__ int frag_k(int lane, int i) {
  int v = i >> 1;
  return ((v & 3) << 1) | ((v >> 2) << 4) | ((lane >> 4) << 3) | (i & 1);
}

__device__ __forceinline__ v8f wmma_f16(v16h a, v16h b, v8f c) {
  return __builtin_amdgcn_wmma_f32_16x16x32_f16(false, a, false, b, (short)0, c,
                                                false, false);
}

// ---------------------------------------------------------------------------
// Layout transforms in/out:  x (T,C,S) <-> xt (N=T*S, C)
// ---------------------------------------------------------------------------
__global__ void k_transpose_in(const float* __restrict__ x,
                               const float* __restrict__ y,
                               float* __restrict__ xt, float* __restrict__ yt) {
  int tid = blockIdx.x * blockDim.x + threadIdx.x;
  if (tid >= NN * CC) return;
  int c = tid % CC, n = tid / CC;
  int t = n / SS, s = n % SS;
  size_t src = ((size_t)t * CC + c) * SS + s;
  xt[tid] = x[src];
  yt[tid] = y[src];
}

__global__ void k_transpose_out(const float* __restrict__ xt,
                                float* __restrict__ out) {
  int tid = blockIdx.x * blockDim.x + threadIdx.x;
  if (tid >= NN * CC) return;
  int s = tid % SS;
  int c = (tid / SS) % CC;
  int t = tid / (SS * CC);
  out[tid] = xt[((size_t)t * SS + s) * CC + c];
}

// ---------------------------------------------------------------------------
// LayerNorm (+ optional temporal-position add) -> f16 row-major.
// One wave per row of C=512 (16 elements / lane), shuffle reductions.
// ---------------------------------------------------------------------------
__global__ void k_ln(const float* __restrict__ src, const float* __restrict__ g,
                     const float* __restrict__ bb, const float* __restrict__ pos,
                     h16* __restrict__ dst) {
  int wid  = (blockIdx.x * blockDim.x + threadIdx.x) >> 5;
  int lane = threadIdx.x & 31;
  if (wid >= NN) return;
  const float* r = src + (size_t)wid * CC;
  int t = wid / SS;
  float vals[16];
  float s = 0.f;
#pragma unroll
  for (int j = 0; j < 16; ++j) { vals[j] = r[lane + 32 * j]; s += vals[j]; }
#pragma unroll
  for (int o = 1; o < 32; o <<= 1) s += __shfl_xor(s, o);
  float mean = s * (1.0f / CC);
  float vs = 0.f;
#pragma unroll
  for (int j = 0; j < 16; ++j) { float d = vals[j] - mean; vs += d * d; }
#pragma unroll
  for (int o = 1; o < 32; o <<= 1) vs += __shfl_xor(vs, o);
  float inv = rsqrtf(vs * (1.0f / CC) + 1e-5f);
#pragma unroll
  for (int j = 0; j < 16; ++j) {
    int c = lane + 32 * j;
    float o = (vals[j] - mean) * inv * g[c] + bb[c];
    if (pos) o += pos[(size_t)t * CC + c];
    dst[(size_t)wid * CC + c] = (h16)o;
  }
}

// ---------------------------------------------------------------------------
// Fragment packers.  Packed A: [mt][kt][lane][16]; Packed B: [kt][nt][lane][16]
// ---------------------------------------------------------------------------
__global__ void k_pack_a_f16(const h16* __restrict__ src, h16* __restrict__ dst,
                             int M, int K) {
  size_t tid = (size_t)blockIdx.x * blockDim.x + threadIdx.x;
  if (tid >= (size_t)M * K) return;
  int i = tid & 15, lane = (tid >> 4) & 31;
  size_t rest = tid >> 9;
  int Kt = K >> 5;
  int kt = (int)(rest % Kt), mt = (int)(rest / Kt);
  int m = mt * 16 + (lane & 15);
  int k = kt * 32 + frag_k(lane, i);
  dst[tid] = src[(size_t)m * K + k];
}

__global__ void k_pack_b_f32(const float* __restrict__ src, h16* __restrict__ dst,
                             int K, int Nn) {
  size_t tid = (size_t)blockIdx.x * blockDim.x + threadIdx.x;
  if (tid >= (size_t)K * Nn) return;
  int i = tid & 15, lane = (tid >> 4) & 31;
  size_t rest = tid >> 9;
  int Nt = Nn >> 4;
  int nt = (int)(rest % Nt), kt = (int)(rest / Nt);
  int n = nt * 16 + (lane & 15);
  int k = kt * 32 + frag_k(lane, i);
  dst[tid] = (h16)src[(size_t)k * Nn + n];
}

// ---------------------------------------------------------------------------
// WMMA GEMM: D(MxNn) = Apk(MxK) * Bpk(KxNn) [+bias] [epilogue by mode]
// mode 0: f32 store   mode 1: f32 store + bias
// mode 2: +bias, GELU, f16 store     mode 3: +bias + resid, f32 store
// One wave computes a 32x64 tile: each B-fragment feeds two WMMAs
// (8 WMMAs per 12 b128 loads per K-step).
// ---------------------------------------------------------------------------
__global__ void k_gemm(const h16* __restrict__ Apk, const h16* __restrict__ Bpk,
                       float* __restrict__ D, h16* __restrict__ Dh,
                       const float* __restrict__ bias,
                       const float* __restrict__ resid,
                       int M, int K, int Nn, int mode) {
  int wid  = (blockIdx.x * blockDim.x + threadIdx.x) >> 5;
  int lane = threadIdx.x & 31;
  int nb   = Nn >> 6;
  int mt2 = wid / nb, nb4 = wid % nb;
  if (mt2 >= (M >> 5)) return;
  int Kt = K >> 5, Nt = Nn >> 4;
  v8f acc[2][4] = {};
  const v16h* Ap0 = (const v16h*)Apk + (size_t)(mt2 * 2) * Kt * 32 + lane;
  const v16h* Ap1 = Ap0 + (size_t)Kt * 32;
  for (int kt = 0; kt < Kt; ++kt) {
    v16h a0 = Ap0[(size_t)kt * 32];
    v16h a1 = Ap1[(size_t)kt * 32];
    const v16h* Bp = (const v16h*)Bpk + ((size_t)kt * Nt + nb4 * 4) * 32 + lane;
#pragma unroll
    for (int t = 0; t < 4; ++t) {
      v16h b = Bp[(size_t)t * 32];
      acc[0][t] = wmma_f16(a0, b, acc[0][t]);
      acc[1][t] = wmma_f16(a1, b, acc[1][t]);
    }
  }
  int halfid = lane >> 4, col = lane & 15;
#pragma unroll
  for (int r = 0; r < 2; ++r) {
#pragma unroll
    for (int t = 0; t < 4; ++t) {
      int n = (nb4 * 4 + t) * 16 + col;
      float bv = (mode >= 1) ? bias[n] : 0.f;
#pragma unroll
      for (int v = 0; v < 8; ++v) {
        int m = mt2 * 32 + r * 16 + v + halfid * 8;
        float x = acc[r][t][v] + bv;
        size_t idx = (size_t)m * Nn + n;
        if (mode == 2) {
          float x3 = x * x * x;
          float gl = 0.5f * x * (1.f + tanhf(0.7978845608f * (x + 0.044715f * x3)));
          Dh[idx] = (h16)gl;
        } else if (mode == 3) {
          D[idx] = x + resid[idx];
        } else {
          D[idx] = x;
        }
      }
    }
  }
}

// ---------------------------------------------------------------------------
// RoPE value for row-major (N x INNER) activations, head col0 = h*64
// ---------------------------------------------------------------------------
__device__ __forceinline__ float rope_val(const float* __restrict__ base,
                                          int n, int col0, int d) {
  float xv = base[(size_t)n * INNER + col0 + d];
  int j = d & 31;
  float invf = __expf(-(float)j * 0.28782313f);  // log(10000)/32
  float fr = (float)n * invf;
  float cs = cosf(fr), sn = sinf(fr);
  float other = (d < 32) ? -base[(size_t)n * INNER + col0 + d + 32]
                         :  base[(size_t)n * INNER + col0 + d - 32];
  return xv * cs + other * sn;
}

// Q: per-head A-frag layout [h][mt][kt(2)][lane][16], RoPE fused
__global__ void k_rope_pack_q(const float* __restrict__ qf, h16* __restrict__ dst) {
  size_t tid = (size_t)blockIdx.x * blockDim.x + threadIdx.x;
  if (tid >= (size_t)NN * INNER) return;
  int i = tid & 15, lane = (tid >> 4) & 31;
  size_t rest = tid >> 9;
  int kt = (int)(rest & 1); rest >>= 1;
  int mt = (int)(rest % NTQ); int h = (int)(rest / NTQ);
  int n = mt * 16 + (lane & 15);
  int d = kt * 32 + frag_k(lane, i);
  dst[tid] = (h16)rope_val(qf, n, h * DH, d);
}

// K^T: per-head B-frag layout [h][kt(2)][nt(144)][lane][16], RoPE fused
__global__ void k_rope_pack_k(const float* __restrict__ kf, h16* __restrict__ dst) {
  size_t tid = (size_t)blockIdx.x * blockDim.x + threadIdx.x;
  if (tid >= (size_t)NN * INNER) return;
  int i = tid & 15, lane = (tid >> 4) & 31;
  size_t rest = tid >> 9;
  int nt = (int)(rest % NTQ); rest /= NTQ;
  int kt = (int)(rest & 1); int h = (int)(rest >> 1);
  int n = nt * 16 + (lane & 15);
  int d = kt * 32 + frag_k(lane, i);
  dst[tid] = (h16)rope_val(kf, n, h * DH, d);
}

// V: per-head B-frag layout [h][ktile(72)][nt(4)][lane][16]
__global__ void k_pack_v(const float* __restrict__ vf, h16* __restrict__ dst) {
  size_t tid = (size_t)blockIdx.x * blockDim.x + threadIdx.x;
  if (tid >= (size_t)NN * INNER) return;
  int i = tid & 15, lane = (tid >> 4) & 31;
  size_t rest = tid >> 9;
  int nt = (int)(rest & 3); rest >>= 2;
  int ktile = (int)(rest % 72); int h = (int)(rest / 72);
  int kkey = ktile * 32 + frag_k(lane, i);
  int d = nt * 16 + (lane & 15);
  dst[tid] = (h16)vf[(size_t)kkey * INNER + h * DH + d];
}

// ---------------------------------------------------------------------------
// Flash attention: one wave = (head h, 16-query tile mt). Streams keys in
// blocks of 32. Scores via WMMA, online softmax with shfl row reductions,
// P staged via LDS to rebuild an A-fragment, P*V via WMMA.
// ---------------------------------------------------------------------------
__global__ void k_attn(const h16* __restrict__ qpk, const h16* __restrict__ kpk,
                       const h16* __restrict__ vpk, const int* __restrict__ mask,
                       h16* __restrict__ outh) {
  __shared__ h16 lds[4][16 * 32];
  int lane = threadIdx.x & 31;
  int wloc = threadIdx.x >> 5;
  int wid  = (blockIdx.x * blockDim.x + threadIdx.x) >> 5;
  int mt = wid % NTQ, h = wid / NTQ;
  if (h >= NHEADS) return;
  int halfid = lane >> 4, col = lane & 15;

  const v16h* qp = (const v16h*)qpk + (size_t)(h * NTQ + mt) * 2 * 32 + lane;
  v16h aq0 = qp[0];
  v16h aq1 = qp[32];
  const v16h* kb = (const v16h*)kpk + (size_t)h * 2 * NTQ * 32 + lane;
  const v16h* vb = (const v16h*)vpk + (size_t)h * 72 * 4 * 32 + lane;

  float rmax[8], rsum[8];
  v8f Ov[4] = {};
#pragma unroll
  for (int v = 0; v < 8; ++v) { rmax[v] = -1e30f; rsum[v] = 0.f; }
  int tq[8];
#pragma unroll
  for (int v = 0; v < 8; ++v) tq[v] = (mt * 16 + v + halfid * 8) / SS;

  h16* P = lds[wloc];

  for (int jp = 0; jp < 72; ++jp) {
    int j0 = jp * 2, j1 = j0 + 1;
    v8f s0 = {}, s1 = {};
    s0 = wmma_f16(aq0, kb[(size_t)(0 * NTQ + j0) * 32], s0);
    s0 = wmma_f16(aq1, kb[(size_t)(1 * NTQ + j0) * 32], s0);
    s1 = wmma_f16(aq0, kb[(size_t)(0 * NTQ + j1) * 32], s1);
    s1 = wmma_f16(aq1, kb[(size_t)(1 * NTQ + j1) * 32], s1);
    int tk0 = (j0 * 16 + col) / SS;
    int tk1 = (j1 * 16 + col) / SS;
#pragma unroll
    for (int v = 0; v < 8; ++v) {
      float x0 = s0[v] * ATT_SCALE;
      float x1 = s1[v] * ATT_SCALE;
      if (mask[tq[v] * TT + tk0] == 0) x0 = -1e30f;
      if (mask[tq[v] * TT + tk1] == 0) x1 = -1e30f;
      float rm = fmaxf(x0, x1);
      rm = fmaxf(rm, __shfl_xor(rm, 1));
      rm = fmaxf(rm, __shfl_xor(rm, 2));
      rm = fmaxf(rm, __shfl_xor(rm, 4));
      rm = fmaxf(rm, __shfl_xor(rm, 8));
      float nm = fmaxf(rmax[v], rm);
      float corr = __expf(rmax[v] - nm);
      float p0 = __expf(x0 - nm);
      float p1 = __expf(x1 - nm);
      float ps = p0 + p1;
      ps += __shfl_xor(ps, 1);
      ps += __shfl_xor(ps, 2);
      ps += __shfl_xor(ps, 4);
      ps += __shfl_xor(ps, 8);
      rsum[v] = rsum[v] * corr + ps;
      rmax[v] = nm;
#pragma unroll
      for (int t = 0; t < 4; ++t) Ov[t][v] *= corr;
      int row = v + halfid * 8;
      P[row * 32 + col]      = (h16)p0;
      P[row * 32 + 16 + col] = (h16)p1;
    }
    // Rebuild 16x32 A fragment of P from LDS (wave-local, DS ops in order)
    union { v16h hv; unsigned u[8]; } pa;
#pragma unroll
    for (int v = 0; v < 8; ++v) {
      int k0 = ((v & 3) << 1) | ((v >> 2) << 4) | (halfid << 3);
      pa.u[v] = *(const unsigned*)&P[col * 32 + k0];
    }
#pragma unroll
    for (int t = 0; t < 4; ++t)
      Ov[t] = wmma_f16(pa.hv, vb[(size_t)(jp * 4 + t) * 32], Ov[t]);
  }

#pragma unroll
  for (int t = 0; t < 4; ++t) {
#pragma unroll
    for (int v = 0; v < 8; ++v) {
      int m = mt * 16 + v + halfid * 8;
      int d = t * 16 + col;
      outh[(size_t)m * INNER + h * DH + d] = (h16)(Ov[t][v] / rsum[v]);
    }
  }
}

// ---------------------------------------------------------------------------
// Gate + residual:  xt[n,:] += proj[n,:] * sigmoid(dot(proj[n,:], gw) + gb)
// One wave per row.
// ---------------------------------------------------------------------------
__global__ void k_gate_residual(const float* __restrict__ proj,
                                const float* __restrict__ gw,
                                const float* __restrict__ gb,
                                float* __restrict__ xt) {
  int wid  = (blockIdx.x * blockDim.x + threadIdx.x) >> 5;
  int lane = threadIdx.x & 31;
  if (wid >= NN) return;
  const float* r = proj + (size_t)wid * CC;
  float s = 0.f;
#pragma unroll
  for (int j = 0; j < 16; ++j) s += r[lane + 32 * j] * gw[lane + 32 * j];
#pragma unroll
  for (int o = 1; o < 32; o <<= 1) s += __shfl_xor(s, o);
  float gl = 1.f / (1.f + __expf(-(s + gb[0])));
#pragma unroll
  for (int j = 0; j < 16; ++j) {
    size_t idx = (size_t)wid * CC + lane + 32 * j;
    xt[idx] += proj[idx] * gl;
  }
}

// ---------------------------------------------------------------------------
// Host orchestration
// ---------------------------------------------------------------------------
extern "C" void kernel_launch(void* const* d_in, const int* in_sizes, int n_in,
                              void* d_out, int out_size, void* d_ws, size_t ws_size,
                              hipStream_t stream) {
  (void)in_sizes; (void)n_in; (void)out_size; (void)ws_size;
  const float* x_in   = (const float*)d_in[0];
  const float* y_in   = (const float*)d_in[1];
  const int*   tmask  = (const int*)d_in[2];
  const float* normxg = (const float*)d_in[3];
  const float* normxb = (const float*)d_in[4];
  const float* normyg = (const float*)d_in[5];
  const float* normyb = (const float*)d_in[6];
  const float* wq     = (const float*)d_in[7];
  const float* wk     = (const float*)d_in[8];
  const float* wv     = (const float*)d_in[9];
  const float* tpos   = (const float*)d_in[10];
  const float* gatew  = (const float*)d_in[11];
  const float* gateb  = (const float*)d_in[12];
  const float* outw   = (const float*)d_in[13];
  const float* outb   = (const float*)d_in[14];
  const float* ffng   = (const float*)d_in[15];
  const float* ffnb   = (const float*)d_in[16];
  const float* w1     = (const float*)d_in[17];
  const float* b1     = (const float*)d_in[18];
  const float* w2     = (const float*)d_in[19];
  const float* b2     = (const float*)d_in[20];
  float* outp = (float*)d_out;

  // ---- workspace carve-up (~100 MB) ----
  char* wsb = (char*)d_ws;
  size_t off = 0;
  auto alloc = [&](size_t bytes) -> void* {
    void* p = wsb + off;
    off += (bytes + 255) & ~(size_t)255;
    return p;
  };
  float* xt    = (float*)alloc((size_t)NN * CC * 4);
  float* yt    = (float*)alloc((size_t)NN * CC * 4);
  h16*   xn_h  = (h16*)  alloc((size_t)NN * CC * 2);
  h16*   yn_h  = (h16*)  alloc((size_t)NN * CC * 2);
  h16*   apkX  = (h16*)  alloc((size_t)NN * CC * 2);
  h16*   apkY  = (h16*)  alloc((size_t)NN * CC * 2);
  h16*   wpk0  = (h16*)  alloc((size_t)CC * MLP * 2);
  h16*   wpk1  = (h16*)  alloc((size_t)CC * MLP * 2);
  h16*   wpk2  = (h16*)  alloc((size_t)CC * MLP * 2);
  float* qf    = (float*)alloc((size_t)NN * INNER * 4);
  float* kf    = (float*)alloc((size_t)NN * INNER * 4);
  float* vf    = (float*)alloc((size_t)NN * INNER * 4);
  h16*   qpk   = (h16*)  alloc((size_t)NN * INNER * 2);
  h16*   kpk   = (h16*)  alloc((size_t)NN * INNER * 2);
  h16*   vpk   = (h16*)  alloc((size_t)NN * INNER * 2);
  h16*   aout  = (h16*)  alloc((size_t)NN * INNER * 2);
  h16*   apkO  = (h16*)  alloc((size_t)NN * INNER * 2);
  float* projf = (float*)alloc((size_t)NN * CC * 4);
  h16*   h1_h  = (h16*)  alloc((size_t)NN * MLP * 2);
  h16*   apkH  = (h16*)  alloc((size_t)NN * MLP * 2);

  auto blocks_for = [](size_t total, int bs) { return (int)((total + bs - 1) / bs); };
  auto gemm = [&](const h16* A, const h16* Bp, float* D, h16* Dh,
                  const float* bias, const float* resid, int M, int K, int Nn,
                  int mode) {
    int waves = (M / 32) * (Nn / 64);
    k_gemm<<<blocks_for((size_t)waves * 32, 256), 256, 0, stream>>>(
        A, Bp, D, Dh, bias, resid, M, K, Nn, mode);
  };
  auto packA = [&](const h16* src, h16* dst, int M, int K) {
    k_pack_a_f16<<<blocks_for((size_t)M * K, 256), 256, 0, stream>>>(src, dst, M, K);
  };
  auto packB = [&](const float* src, h16* dst, int K, int Nn) {
    k_pack_b_f32<<<blocks_for((size_t)K * Nn, 256), 256, 0, stream>>>(src, dst, K, Nn);
  };

  k_transpose_in<<<blocks_for((size_t)NN * CC, 256), 256, 0, stream>>>(x_in, y_in, xt, yt);

  for (int i = 0; i < 2; ++i) {
    const float* pos = tpos + (size_t)i * MAXT * CC;
    // --- LN + pos -> f16, then pack as WMMA A operands ---
    k_ln<<<NN / 8, 256, 0, stream>>>(xt, normxg + i * CC, normxb + i * CC, pos, xn_h);
    k_ln<<<NN / 8, 256, 0, stream>>>(yt, normyg + i * CC, normyb + i * CC, pos, yn_h);
    packA(xn_h, apkX, NN, CC);
    packA(yn_h, apkY, NN, CC);
    // --- QKV GEMMs ---
    packB(wq + (size_t)i * CC * INNER, wpk0, CC, INNER);
    packB(wk + (size_t)i * CC * INNER, wpk1, CC, INNER);
    packB(wv + (size_t)i * CC * INNER, wpk2, CC, INNER);
    gemm(apkX, wpk0, qf, nullptr, nullptr, nullptr, NN, CC, INNER, 0);
    gemm(apkX, wpk1, kf, nullptr, nullptr, nullptr, NN, CC, INNER, 0);
    gemm(apkY, wpk2, vf, nullptr, nullptr, nullptr, NN, CC, INNER, 0);
    // --- RoPE + repack into per-head fragment layouts ---
    k_rope_pack_q<<<blocks_for((size_t)NN * INNER, 256), 256, 0, stream>>>(qf, qpk);
    k_rope_pack_k<<<blocks_for((size_t)NN * INNER, 256), 256, 0, stream>>>(kf, kpk);
    k_pack_v<<<blocks_for((size_t)NN * INNER, 256), 256, 0, stream>>>(vf, vpk);
    // --- Flash attention ---
    k_attn<<<(NHEADS * NTQ) / 4, 128, 0, stream>>>(qpk, kpk, vpk, tmask, aout);
    // --- Output projection + gate + residual ---
    packA(aout, apkO, NN, INNER);
    packB(outw + (size_t)i * INNER * CC, wpk0, INNER, CC);
    gemm(apkO, wpk0, projf, nullptr, outb + i * CC, nullptr, NN, INNER, CC, 1);
    k_gate_residual<<<NN / 8, 256, 0, stream>>>(projf, gatew + i * CC, gateb + i, xt);
    // --- FFN ---
    k_ln<<<NN / 8, 256, 0, stream>>>(xt, ffng + i * CC, ffnb + i * CC, nullptr, xn_h);
    packA(xn_h, apkX, NN, CC);
    packB(w1 + (size_t)i * CC * MLP, wpk0, CC, MLP);
    gemm(apkX, wpk0, nullptr, h1_h, b1 + i * MLP, nullptr, NN, CC, MLP, 2);
    packA(h1_h, apkH, NN, MLP);
    packB(w2 + (size_t)i * MLP * CC, wpk1, MLP, CC);
    gemm(apkH, wpk1, xt, nullptr, b2 + i * CC, xt, NN, MLP, CC, 3);
  }

  k_transpose_out<<<blocks_for((size_t)NN * CC, 256), 256, 0, stream>>>(xt, outp);
}
